// FeatureEmb_6107443495191
// MI455X (gfx1250) — matrix operand order repaired and checked by reading, not compiled, not run on hardware
//
#include <hip/hip_runtime.h>
#include <stdint.h>

// Problem constants (from the reference)
#define TN_      2000
#define UN_      50000
#define TF_      16
#define OUTW_    52          // 16 static + 36 emb
#define NDIM_    48          // 36 emb dims + 1 "ones" col (idx 36) + 11 zero pad
#define NCHUNKS  1563        // ceil(50000 / 32); chunk 1562 is partial (16 users)
#define NFULL    1562        // full 32-user chunks
#define KSPLIT   16
#define MTILES   125         // 2000 / 16
#define CPS      98          // ceil(NFULL / KSPLIT); seg 15 gets 92 + the tail

typedef __attribute__((ext_vector_type(16))) _Float16 v16h;
typedef __attribute__((ext_vector_type(8)))  float    v8f;

union HV  { v16h h; uint32_t w[8]; uint4 q[2]; };
union H16 { _Float16 f; uint16_t b; };

struct Tile { HV a, b0, b1, b2; };

__device__ __forceinline__ uint32_t pack01(int a, int b) {
  // mask entries are 0/1; f16(1.0) == 0x3C00, exact
  return ((a == 1) ? 0x3C00u : 0u) | ((b == 1) ? 0x3C000000u : 0u);
}

// ---------------------------------------------------------------------------
// Kernel 1: gather embeddings -> f16, stored directly in WMMA B-matrix layout.
//   dword index = chunk*768 + nt*256 + lane*8 + j , half = hi
//   lane = (d%16) + 16*((p>>3)&1), j = ((p&7)>>1) + 4*(p>>4), hi = p&1   (p = user slot)
// Column 36 holds 1.0 for valid users (GEMM then produces counts for free).
// ---------------------------------------------------------------------------
__global__ __launch_bounds__(256) void build_B_kernel(
    const int* __restrict__ U,
    const float* __restrict__ e0, const float* __restrict__ e1,
    const float* __restrict__ e2, const float* __restrict__ e3,
    const float* __restrict__ e4, const float* __restrict__ e5,
    uint16_t* __restrict__ wsB) {
  int u = blockIdx.x * blockDim.x + threadIdx.x;
  if (u >= NCHUNKS * 32) return;
  int chunk  = u >> 5;
  int p      = u & 31;
  int laneHi = (p & 8) ? 16 : 0;
  int j      = ((p & 7) >> 1) + 4 * (p >> 4);
  int hi     = p & 1;

  float vals[NDIM_];
  #pragma unroll
  for (int d = 0; d < NDIM_; ++d) vals[d] = 0.0f;

  if (u < UN_) {
    const float* tabs[6] = {e0, e1, e2, e3, e4, e5};
    #pragma unroll
    for (int i = 0; i < 6; ++i) {
      int idx = U[u * 6 + i];
      const float* row = tabs[i] + (long)idx * 6;
      #pragma unroll
      for (int k = 0; k < 6; ++k) vals[i * 6 + k] = row[k];
    }
    vals[36] = 1.0f;  // counts column
  }

  long base = (long)chunk * 768;
  #pragma unroll
  for (int d = 0; d < NDIM_; ++d) {
    int nt = d >> 4, n = d & 15;
    long dw = base + nt * 256 + (n + laneHi) * 8 + j;
    H16 h; h.f = (_Float16)vals[d];
    wsB[dw * 2 + hi] = h.b;
  }
}

// ---------------------------------------------------------------------------
// GEMM helpers
// ---------------------------------------------------------------------------
__device__ __forceinline__ void load_tile(Tile& T, int c,
                                          const int* __restrict__ maskRow,
                                          const uint32_t* __restrict__ wsB,
                                          int lane, int half) {
  const int u0 = c * 32;
  // A tile: 16-bit A 16x32 lane layout (lanes<16: K 0-7/16-23; lanes>=16: +8)
  const int* p0 = maskRow + u0 + 8 * half;            // 16B aligned
  if (u0 + 32 * 9 <= UN_) __builtin_prefetch(p0 + 256, 0, 1);
  int4 m0 = *(const int4*)(p0);
  int4 m1 = *(const int4*)(p0 + 4);
  int4 m2 = *(const int4*)(p0 + 16);
  int4 m3 = *(const int4*)(p0 + 20);
  T.a.w[0] = pack01(m0.x, m0.y); T.a.w[1] = pack01(m0.z, m0.w);
  T.a.w[2] = pack01(m1.x, m1.y); T.a.w[3] = pack01(m1.z, m1.w);
  T.a.w[4] = pack01(m2.x, m2.y); T.a.w[5] = pack01(m2.z, m2.w);
  T.a.w[6] = pack01(m3.x, m3.y); T.a.w[7] = pack01(m3.z, m3.w);
  // B tiles: pre-staged in WMMA layout, lane-contiguous 32B per N-tile
  const uint32_t* bp = wsB + (long)c * 768 + lane * 8;
  T.b0.q[0] = ((const uint4*)(bp      ))[0]; T.b0.q[1] = ((const uint4*)(bp      ))[1];
  T.b1.q[0] = ((const uint4*)(bp + 256))[0]; T.b1.q[1] = ((const uint4*)(bp + 256))[1];
  T.b2.q[0] = ((const uint4*)(bp + 512))[0]; T.b2.q[1] = ((const uint4*)(bp + 512))[1];
}

__device__ __forceinline__ void mma3(const Tile& T, v8f& c0, v8f& c1, v8f& c2) {
  c0 = __builtin_amdgcn_wmma_f32_16x16x32_f16(false, T.a.h, false, T.b0.h,
                                              (short)0, c0, false, false);
  c1 = __builtin_amdgcn_wmma_f32_16x16x32_f16(false, T.a.h, false, T.b1.h,
                                              (short)0, c1, false, false);
  c2 = __builtin_amdgcn_wmma_f32_16x16x32_f16(false, T.a.h, false, T.b2.h,
                                              (short)0, c2, false, false);
}

// ---------------------------------------------------------------------------
// Kernel 2: WMMA GEMM  partial[seg][t, 0..47] = sum_{u in seg} mask[t,u]*B[u,:]
// One wave per (16-team M-tile) x (K segment). Ping-pong double buffering so
// the next chunk's loads are in flight during the current chunk's WMMAs.
// The single ragged chunk (users 49984..49999) is peeled to segment 15.
// ---------------------------------------------------------------------------
__global__ __launch_bounds__(32) void gemm_kernel(
    const int* __restrict__ mask,
    const uint32_t* __restrict__ wsB,
    float* __restrict__ part) {
  const int lane  = threadIdx.x;           // 0..31
  const int mtile = blockIdx.x;            // 0..124
  const int seg   = blockIdx.y;            // 0..KSPLIT-1
  const int t0    = mtile * 16;
  const int half  = lane >> 4;             // 0 | 1
  const int mrow  = lane & 15;             // team row within tile (both halves)
  const int* maskRow = mask + (long)(t0 + mrow) * UN_;

  v8f c0 = {}, c1 = {}, c2 = {};

  const int cbeg = seg * CPS;
  const int cend = (cbeg + CPS < NFULL) ? (cbeg + CPS) : NFULL;

  // ---- main loop over full chunks, software ping-pong ----
  Tile P, Q;
  load_tile(P, cbeg, maskRow, wsB, lane, half);
  int c = cbeg;
  while (c + 1 < cend) {
    load_tile(Q, c + 1, maskRow, wsB, lane, half);  // in flight during P's MMAs
    mma3(P, c0, c1, c2);
    if (c + 2 < cend) load_tile(P, c + 2, maskRow, wsB, lane, half);
    mma3(Q, c0, c1, c2);
    c += 2;
  }
  if (c < cend) mma3(P, c0, c1, c2);

  // ---- peeled ragged tail: chunk 1562 (16 valid users), last segment only ----
  if (seg == KSPLIT - 1) {
    const int u0 = NFULL * 32;             // 49984
    Tile T;
    #pragma unroll
    for (int i = 0; i < 8; ++i) {
      int k0 = ((i < 4) ? (2 * i) : (16 + 2 * (i - 4))) + 8 * half;
      int ua = u0 + k0, ub = u0 + k0 + 1;
      int ma = (ua < UN_) ? maskRow[ua] : 0;
      int mb = (ub < UN_) ? maskRow[ub] : 0;
      T.a.w[i] = pack01(ma, mb);
    }
    const uint32_t* bp = wsB + (long)NFULL * 768 + lane * 8;
    T.b0.q[0] = ((const uint4*)(bp      ))[0]; T.b0.q[1] = ((const uint4*)(bp      ))[1];
    T.b1.q[0] = ((const uint4*)(bp + 256))[0]; T.b1.q[1] = ((const uint4*)(bp + 256))[1];
    T.b2.q[0] = ((const uint4*)(bp + 512))[0]; T.b2.q[1] = ((const uint4*)(bp + 512))[1];
    mma3(T, c0, c1, c2);
  }

  // ---- store partials (C layout: VGPR r -> row r + 8*half, col lane&15) ----
  float* dst = part + ((long)seg * TN_ + t0) * NDIM_;
  const int col = lane & 15;
  #pragma unroll
  for (int r = 0; r < 8; ++r) {
    int row = r + 8 * half;
    dst[(long)row * NDIM_ +  0 + col] = c0[r];
    dst[(long)row * NDIM_ + 16 + col] = c1[r];
    dst[(long)row * NDIM_ + 32 + col] = c2[r];
  }
}

// ---------------------------------------------------------------------------
// Kernel 3: deterministic reduction over K segments, divide by counts (col 36),
// concat T_static -> out [2000 x 52]
// ---------------------------------------------------------------------------
__global__ __launch_bounds__(256) void finalize_kernel(
    const float* __restrict__ Ts,
    const float* __restrict__ part,
    float* __restrict__ out) {
  int idx = blockIdx.x * blockDim.x + threadIdx.x;
  if (idx >= TN_ * OUTW_) return;
  int t = idx / OUTW_, k = idx % OUTW_;
  if (k < TF_) {
    out[idx] = Ts[t * TF_ + k];
  } else {
    int d = k - TF_;
    float s = 0.0f, cnt = 0.0f;
    #pragma unroll
    for (int sg = 0; sg < KSPLIT; ++sg) {
      const float* p = part + ((long)sg * TN_ + t) * NDIM_;
      s   += p[d];
      cnt += p[36];
    }
    out[idx] = s / cnt;
  }
}

// ---------------------------------------------------------------------------
extern "C" void kernel_launch(void* const* d_in, const int* in_sizes, int n_in,
                              void* d_out, int out_size, void* d_ws, size_t ws_size,
                              hipStream_t stream) {
  const float* T_static = (const float*)d_in[0];
  const int*   U_static = (const int*)d_in[1];
  const int*   mask     = (const int*)d_in[2];
  const float* e0 = (const float*)d_in[3];
  const float* e1 = (const float*)d_in[4];
  const float* e2 = (const float*)d_in[5];
  const float* e3 = (const float*)d_in[6];
  const float* e4 = (const float*)d_in[7];
  const float* e5 = (const float*)d_in[8];
  float* out = (float*)d_out;

  // workspace: [B-staging: NCHUNKS*768 dwords][partials: KSPLIT*TN_*48 f32]
  const size_t wsB_bytes = (size_t)NCHUNKS * 768 * 4;   // 4,801,536 (256-aligned)
  uint16_t* wsB  = (uint16_t*)d_ws;
  float*    part = (float*)((char*)d_ws + wsB_bytes);

  // 1) stage embeddings as f16 in WMMA B layout (also zeroes pad users/cols)
  {
    int n = NCHUNKS * 32;
    build_B_kernel<<<(n + 255) / 256, 256, 0, stream>>>(
        U_static, e0, e1, e2, e3, e4, e5, wsB);
  }
  // 2) WMMA GEMM, K split 16 ways (2000 single-wave workgroups)
  {
    dim3 grid(MTILES, KSPLIT);
    gemm_kernel<<<grid, 32, 0, stream>>>(mask, (const uint32_t*)wsB, part);
  }
  // 3) reduce partials, divide by counts, concat static features
  {
    int n = TN_ * OUTW_;
    finalize_kernel<<<(n + 255) / 256, 256, 0, stream>>>(T_static, part, out);
  }
}
